// Quantizer_80668075754205
// MI455X (gfx1250) — matrix operand order repaired and checked by reading, not compiled, not run on hardware
//
#include <hip/hip_runtime.h>
#include <hip/hip_bf16.h>
#include <stdint.h>

// ---------------------------------------------------------------------------
// VQ quantizer for MI455X (gfx1250): bf16 WMMA distance GEMM + argmin,
// async-to-LDS codebook staging, exact f32 gather + loss.
// ---------------------------------------------------------------------------

#define N_PTS 32768
#define DIM   256
#define KCB   2048
#define BETA  0.25f

typedef __bf16 v16bf __attribute__((ext_vector_type(16)));
typedef __bf16 v4bf  __attribute__((ext_vector_type(4)));
typedef float  v8f   __attribute__((ext_vector_type(8)));
typedef float  v4f   __attribute__((ext_vector_type(4)));
typedef unsigned u32x4 __attribute__((ext_vector_type(4)));

union FragBF { v16bf v; u32x4 u[2]; };

// --------------------------- conversion kernels ----------------------------

__global__ void vq_cvt_bf16(const float* __restrict__ src,
                            __bf16* __restrict__ dst, int n4) {
  int i = blockIdx.x * blockDim.x + threadIdx.x;
  if (i < n4) {
    v4f f = ((const v4f*)src)[i];
    ((v4bf*)dst)[i] = __builtin_convertvector(f, v4bf);
  }
}

__global__ void vq_cnorm(const float* __restrict__ cb, float* __restrict__ cn) {
  int k = blockIdx.x * blockDim.x + threadIdx.x;
  if (k < KCB) {
    const float* row = cb + (size_t)k * DIM;
    float s = 0.f;
    #pragma unroll 8
    for (int d = 0; d < DIM; ++d) { float v = row[d]; s += v * v; }
    cn[k] = s;
  }
}

// --------------------------- distance + argmin -----------------------------
// Block: 256 threads = 8 waves. Each wave owns 16 rows of x (A fragments in
// registers). Codebook streamed through double-buffered LDS via async loads.

#define TC       128                  // codes per LDS supertile
#define NST      (KCB / TC)           // 16 supertiles
#define PITCHBF  264                  // bf16 elements per LDS row (bank pad)
#define STBYTES  (TC * PITCHBF * 2)   // one LDS buffer

__device__ __forceinline__ void vq_async_tile(__bf16* bufbase,
                                              const __bf16* __restrict__ cbb,
                                              int st, int tid) {
  // copy TC rows x 512B (contiguous) -> LDS rows with 528B pitch
  unsigned lds_base = (unsigned)(uintptr_t)bufbase;   // low 32b = LDS offset
  const char* gsrc = (const char*)(cbb + (size_t)st * TC * DIM);
  #pragma unroll
  for (int r = 0; r < (TC * DIM * 2) / (256 * 16); ++r) {   // 16 rounds
    unsigned byteidx = (unsigned)tid * 16u + (unsigned)r * 4096u;
    unsigned code    = byteidx >> 9;          // /512
    unsigned within  = byteidx & 511u;
    unsigned lds_off = lds_base + code * (PITCHBF * 2u) + within;
    unsigned long long ga = (unsigned long long)(gsrc + (size_t)code * 512u + within);
    asm volatile("global_load_async_to_lds_b128 %0, %1, off"
                 :: "v"(lds_off), "v"(ga) : "memory");
  }
}

__global__ __launch_bounds__(256, 1)
void vq_dist_argmin(const __bf16* __restrict__ xb,
                    const __bf16* __restrict__ cbb,
                    const float*  __restrict__ cnorm,
                    unsigned*     __restrict__ amin) {
  extern __shared__ __bf16 sB[];                 // [2][TC][PITCHBF]
  const int tid  = threadIdx.x;
  const int wave = tid >> 5;
  const int lane = tid & 31;
  const int h    = lane >> 4;                    // lane half (0/1)
  const int ln   = lane & 15;
  const int rowbase = blockIdx.x * 128 + wave * 16;

  // A fragments: row M = ln; lane half h selects K groups {8h..8h+7, 8h+16..8h+23}
  FragBF a[8];
  const __bf16* xrow = xb + (size_t)(rowbase + ln) * DIM;
  #pragma unroll
  for (int ch = 0; ch < 8; ++ch) {
    int k0 = 32 * ch + 8 * h;
    a[ch].u[0] = *(const u32x4*)(xrow + k0);
    a[ch].u[1] = *(const u32x4*)(xrow + k0 + 16);
  }

  float    bestv[8];
  unsigned besti[8];
  #pragma unroll
  for (int r = 0; r < 8; ++r) { bestv[r] = 3.0e38f; besti[r] = 0u; }

  vq_async_tile(sB, cbb, 0, tid);
  asm volatile("s_wait_asynccnt 0" ::: "memory");
  __syncthreads();

  for (int st = 0; st < NST; ++st) {
    __bf16* buf = sB + (size_t)(st & 1) * (TC * PITCHBF);
    if (st + 1 < NST)
      vq_async_tile(sB + (size_t)((st + 1) & 1) * (TC * PITCHBF), cbb, st + 1, tid);

    #pragma unroll 2
    for (int ct = 0; ct < TC / 16; ++ct) {
      const int cglob = st * TC + ct * 16 + ln;              // this lane's column
      const float cn = cnorm[cglob];                         // issue before LDS burst
      const __bf16* brow = buf + (size_t)(ct * 16 + ln) * PITCHBF;

      // Batch all B fragments: 16 ds_load_b128 issued back-to-back into
      // distinct registers, then a hard scheduling fence so the WMMA stream
      // only needs partial (in-order) dscnt waits.
      FragBF b[8];
      #pragma unroll
      for (int ch = 0; ch < 8; ++ch) {
        int k0 = 32 * ch + 8 * h;                            // K = 32ch+{8h..8h+7, 8h+16..8h+23}
        b[ch].u[0] = *(const u32x4*)(brow + k0);
        b[ch].u[1] = *(const u32x4*)(brow + k0 + 16);
      }
      __builtin_amdgcn_sched_barrier(0);                     // loads before MMAs

      // Two independent accumulator chains (halve D->C dependency depth).
      v8f acc0 = {}, acc1 = {};
      #pragma unroll
      for (int ch = 0; ch < 8; ch += 2) {
        acc0 = __builtin_amdgcn_wmma_f32_16x16x32_bf16(
            false, a[ch].v,     false, b[ch].v,     (short)0, acc0, false, false);
        acc1 = __builtin_amdgcn_wmma_f32_16x16x32_bf16(
            false, a[ch + 1].v, false, b[ch + 1].v, (short)0, acc1, false, false);
      }

      #pragma unroll
      for (int r = 0; r < 8; ++r) {
        float dist = cn - 2.0f * (acc0[r] + acc1[r]);        // row r+8h, col ln
        if (dist < bestv[r]) { bestv[r] = dist; besti[r] = (unsigned)cglob; }
      }
    }
    asm volatile("s_wait_asynccnt 0" ::: "memory");
    __syncthreads();
  }

  // per-row min+index across the 16 column lanes (halves are independent rows)
  #pragma unroll
  for (int r = 0; r < 8; ++r) {
    float v = bestv[r]; unsigned i = besti[r];
    #pragma unroll
    for (int mask = 8; mask >= 1; mask >>= 1) {
      float    ov = __shfl_xor(v, mask, 16);
      unsigned oi = (unsigned)__shfl_xor((int)i, mask, 16);
      if (ov < v || (ov == v && oi < i)) { v = ov; i = oi; }
    }
    if (ln == 0) amin[rowbase + 8 * h + r] = i;              // rows r+8h
  }
}

// --------------------------- gather + loss ---------------------------------

__global__ __launch_bounds__(256)
void vq_gather_loss(const float* __restrict__ x,
                    const float* __restrict__ cb,
                    const unsigned* __restrict__ amin,
                    float* __restrict__ out,
                    float* __restrict__ partials) {
  __shared__ float red[256];
  const int tid = threadIdx.x;
  const int rowbase = blockIdx.x * 8;
  float s = 0.f;
  #pragma unroll
  for (int r = 0; r < 8; ++r) {
    const int row = rowbase + r;
    const unsigned idx = amin[row];
    const float c = cb[(size_t)idx * DIM + tid];
    const float d = x[(size_t)row * DIM + tid] - c;
    out[(size_t)row * DIM + tid] = c;
    s += d * d;
  }
  red[tid] = s;
  __syncthreads();
  for (int off = 128; off > 0; off >>= 1) {
    if (tid < off) red[tid] += red[tid + off];
    __syncthreads();
  }
  if (tid == 0) partials[blockIdx.x] = red[0];
}

__global__ __launch_bounds__(256)
void vq_final_loss(const float* __restrict__ partials, float* __restrict__ loss_out) {
  __shared__ float red[256];
  const int tid = threadIdx.x;
  float s = 0.f;
  for (int i = tid; i < N_PTS / 8; i += 256) s += partials[i];
  red[tid] = s;
  __syncthreads();
  for (int off = 128; off > 0; off >>= 1) {
    if (tid < off) red[tid] += red[tid + off];
    __syncthreads();
  }
  if (tid == 0)
    *loss_out = (1.0f + BETA) * red[0] / (float)((double)N_PTS * (double)DIM);
}

// --------------------------- launch ----------------------------------------

extern "C" void kernel_launch(void* const* d_in, const int* in_sizes, int n_in,
                              void* d_out, int out_size, void* d_ws, size_t ws_size,
                              hipStream_t stream) {
  const float* x  = (const float*)d_in[0];   // [N, D]
  const float* cb = (const float*)d_in[1];   // [K, D]
  float* out = (float*)d_out;                // [N*D] quantized, then [1] loss

  char* ws = (char*)d_ws;
  const size_t SZ_XB  = (size_t)N_PTS * DIM * 2;   // x bf16
  const size_t SZ_CBB = (size_t)KCB * DIM * 2;     // codebook bf16
  const size_t SZ_CN  = (size_t)KCB * 4;           // ||c||^2
  const size_t SZ_AM  = (size_t)N_PTS * 4;         // argmin indices
  __bf16*   xb   = (__bf16*)(ws);
  __bf16*   cbb  = (__bf16*)(ws + SZ_XB);
  float*    cn   = (float*)(ws + SZ_XB + SZ_CBB);
  unsigned* am   = (unsigned*)(ws + SZ_XB + SZ_CBB + SZ_CN);
  float*    ps   = (float*)(ws + SZ_XB + SZ_CBB + SZ_CN + SZ_AM);

  // 1) f32 -> bf16 conversions
  {
    int n4 = N_PTS * DIM / 4;
    vq_cvt_bf16<<<n4 / 256, 256, 0, stream>>>(x, xb, n4);
  }
  {
    int n4 = KCB * DIM / 4;
    vq_cvt_bf16<<<n4 / 256, 256, 0, stream>>>(cb, cbb, n4);
  }
  // 2) exact f32 code norms
  vq_cnorm<<<KCB / 256, 256, 0, stream>>>(cb, cn);

  // 3) WMMA distance GEMM + argmin (double-buffered async LDS staging)
  vq_dist_argmin<<<N_PTS / 128, 256, 2 * STBYTES, stream>>>(xb, cbb, cn, am);

  // 4) gather nearest codes (exact f32) + per-block SSE partials
  vq_gather_loss<<<N_PTS / 8, 256, 0, stream>>>(x, cb, am, out, ps);

  // 5) deterministic final reduce -> loss scalar at out[N*D]
  vq_final_loss<<<1, 256, 0, stream>>>(ps, out + (size_t)N_PTS * DIM);
}
